// SAE_Contrastive_40733469835424
// MI455X (gfx1250) — compile-verified, matrix operand order checked by reading, not run on hardware
//
#include <hip/hip_runtime.h>

// CDNA5 / gfx1250 fused top-K SAE forward.
// Pass 1: bf16 conversion of (x - b_dec) and W_enc into workspace.
// Pass 2: fused bf16 WMMA encoder GEMM (double-buffered B fragments)
//         + running per-row top-32 + decoder gather.

typedef __bf16 bf16_t;
typedef __attribute__((ext_vector_type(16))) __bf16 v16bf;
typedef __attribute__((ext_vector_type(8)))  __bf16 v8bf;
typedef __attribute__((ext_vector_type(8)))  float  v8f;

namespace {
constexpr int B_TOK = 32768;   // tokens
constexpr int D_IN  = 256;     // input dim
constexpr int H_LAT = 8192;    // latents
constexpr int TOPK  = 32;
constexpr int ROWS  = 64;      // rows per workgroup (4 WMMA M tiles)
constexpr int SUB   = 32;      // latent columns per WMMA pass (2 N tiles)
constexpr int SUPER = 64;      // latent columns per barrier/scan phase
}

// ---------------------------------------------------------------------------
// Kernel 1: f32 -> bf16 conversion. xa = x - b_dec (sae_in), wb = W_enc.
// ---------------------------------------------------------------------------
__global__ __launch_bounds__(256) void sae_convert_bf16(
    const float* __restrict__ x, const float* __restrict__ W_enc,
    const float* __restrict__ b_dec,
    bf16_t* __restrict__ xa, bf16_t* __restrict__ wb)
{
  const int stride = gridDim.x * blockDim.x;
  const int i = blockIdx.x * blockDim.x + threadIdx.x;
  for (int t = i; t < B_TOK * D_IN; t += stride)
    xa[t] = (bf16_t)(x[t] - b_dec[t & (D_IN - 1)]);
  for (int t = i; t < H_LAT * D_IN; t += stride)
    wb[t] = (bf16_t)W_enc[t];
}

// Load one 16x32 bf16 fragment k-step: 2 x b128 per lane.
// 16-bit layout: lane r (0..15): K {0..7, 16..23}; lane r+16: K {8..15, 24..31}.
__device__ __forceinline__ v16bf load_frag_step(const bf16_t* p) {
  v8bf lo = *(const v8bf*)(p);
  v8bf hi = *(const v8bf*)(p + 16);
  return __builtin_shufflevector(lo, hi,
           0,1,2,3,4,5,6,7,8,9,10,11,12,13,14,15);
}

__device__ __forceinline__ void load_frags8(v16bf dst[8], const bf16_t* base) {
#pragma unroll
  for (int s = 0; s < 8; ++s) dst[s] = load_frag_step(base + s * 32);
}

// ---------------------------------------------------------------------------
// Kernel 2: fused encoder WMMA GEMM + relu + top-K + decoder gather.
// 256 threads = 8 wave32. Wave w: M tile = w&3 (rows), N tile = w>>2 (cols).
// ---------------------------------------------------------------------------
__global__ __launch_bounds__(256) void sae_fused_topk(
    const bf16_t* __restrict__ xa, const bf16_t* __restrict__ wb,
    const float* __restrict__ b_enc, const float* __restrict__ W_dec,
    const float* __restrict__ b_dec, float* __restrict__ out)
{
  __shared__ float stage[ROWS][SUPER + 1];  // +1 pad to spread LDS banks
  __shared__ float kval[ROWS][TOPK];
  __shared__ int   kidx[ROWS][TOPK];

  const int tid   = threadIdx.x;
  const int lane  = tid & 31;
  const int wave  = tid >> 5;
  const int mTile = wave & 3;   // 0..3 -> rows [mTile*16, +16)
  const int nTile = wave >> 2;  // 0..1 -> cols [nTile*16, +16) within sub-chunk
  const int rowBase = blockIdx.x * ROWS;

  // Init per-row top-K lists. relu() >= 0, so a 0-initialized list means
  // "no contribution" (val 0 * W_dec[0] == 0, matching reference zeros).
  if (tid < ROWS) {
#pragma unroll
    for (int k = 0; k < TOPK; ++k) { kval[tid][k] = 0.f; kidx[tid][k] = 0; }
  }
  float curMin = 0.f;   // running min of this owner-thread's row list
  int   minPos = 0;

  // --- Preload A fragments for this wave's 16-row M tile (all of K=256). ---
  v16bf afrag[8];
  load_frags8(afrag,
      xa + (size_t)(rowBase + mTile * 16 + (lane & 15)) * D_IN + (lane >> 4) * 8);
  __syncthreads();

  const int rBaseLocal = mTile * 16 + (lane >> 4) * 8;  // C/D: VGPR j -> M=j (+8 hi lanes)
  const int cLocal     = nTile * 16 + (lane & 15);      // C/D: lane -> N

  // B fragment base pointer for this wave's N tile; column col is at
  // bbase + col * D_IN (lane n reads row (col+n) of W_enc, contiguous in K).
  const bf16_t* bbase =
      wb + (size_t)(nTile * 16 + (lane & 15)) * D_IN + (lane >> 4) * 8;

  // Double-buffered B fragments: prefetch next sub-chunk while WMMA'ing
  // the current one, so s_wait_loadcnt before the WMMAs only covers loads
  // issued a full phase earlier.
  v16bf bufA[8], bufB[8];
  load_frags8(bufA, bbase);  // columns [0, 32)

  for (int hb = 0; hb < H_LAT; hb += SUPER) {
    // ---- sub-chunk 0: consume bufA (cols hb..hb+31), prefetch bufB ----
    load_frags8(bufB, bbase + (size_t)(hb + SUB) * D_IN);
    {
      const float bias = b_enc[hb + cLocal];
      v8f c = {};
#pragma unroll
      for (int s = 0; s < 8; ++s)
        c = __builtin_amdgcn_wmma_f32_16x16x32_bf16(
                false, afrag[s], false, bufA[s], (short)0, c, false, false);
#pragma unroll
      for (int j = 0; j < 8; ++j) {
        float v = c[j] + bias;
        stage[rBaseLocal + j][cLocal] = v > 0.f ? v : 0.f;
      }
    }

    // ---- sub-chunk 1: consume bufB (cols hb+32..hb+63), prefetch bufA ----
    // Final prefetch wraps to column 0 (valid memory, never consumed).
    const int hn = (hb + SUPER < H_LAT) ? hb + SUPER : 0;
    load_frags8(bufA, bbase + (size_t)hn * D_IN);
    {
      const float bias = b_enc[hb + SUB + cLocal];
      v8f c = {};
#pragma unroll
      for (int s = 0; s < 8; ++s)
        c = __builtin_amdgcn_wmma_f32_16x16x32_bf16(
                false, afrag[s], false, bufB[s], (short)0, c, false, false);
#pragma unroll
      for (int j = 0; j < 8; ++j) {
        float v = c[j] + bias;
        stage[rBaseLocal + j][SUB + cLocal] = v > 0.f ? v : 0.f;
      }
    }
    __syncthreads();

    // ---- Top-K update: one owner thread per row, incremental min tracking.
    if (tid < ROWS) {
#pragma unroll 4
      for (int cc = 0; cc < SUPER; ++cc) {
        float v = stage[tid][cc];
        if (v > curMin) {
          kval[tid][minPos] = v;
          kidx[tid][minPos] = hb + cc;
          curMin = kval[tid][0]; minPos = 0;
#pragma unroll
          for (int k = 1; k < TOPK; ++k) {
            float kv = kval[tid][k];
            if (kv < curMin) { curMin = kv; minPos = k; }
          }
        }
      }
    }
    __syncthreads();
  }

  // --- Decoder gather: out[row] = b_dec + sum_k val_k * W_dec[idx_k]. ---
  // 256 threads cover one row's 256 output columns; kval/kidx reads are
  // uniform per iteration -> LDS broadcast; W_dec loads are coalesced and
  // L2-resident (8.4 MB << 192 MB).
  const float bd = b_dec[tid];
  for (int r = 0; r < ROWS; ++r) {
    float acc = bd;
#pragma unroll 8
    for (int k = 0; k < TOPK; ++k) {
      float v = kval[r][k];
      int   h = kidx[r][k];
      acc = fmaf(v, W_dec[(size_t)h * D_IN + tid], acc);
    }
    out[(size_t)(rowBase + r) * D_IN + tid] = acc;
  }
}

// ---------------------------------------------------------------------------
extern "C" void kernel_launch(void* const* d_in, const int* in_sizes, int n_in,
                              void* d_out, int out_size, void* d_ws, size_t ws_size,
                              hipStream_t stream) {
  (void)in_sizes; (void)n_in; (void)out_size; (void)ws_size;
  const float* x     = (const float*)d_in[0];  // [B, D]
  const float* W_enc = (const float*)d_in[1];  // [H, D]
  const float* b_enc = (const float*)d_in[2];  // [H]
  const float* W_dec = (const float*)d_in[3];  // [H, D]
  const float* b_dec = (const float*)d_in[4];  // [D]
  float* out = (float*)d_out;                  // [B, D] f32

  // Workspace: bf16 sae_in (16.8 MB) + bf16 W_enc (4.2 MB).
  bf16_t* xa = (bf16_t*)d_ws;
  bf16_t* wb = (bf16_t*)((char*)d_ws + (size_t)B_TOK * D_IN * sizeof(bf16_t));

  sae_convert_bf16<<<2048, 256, 0, stream>>>(x, W_enc, b_dec, xa, wb);
  sae_fused_topk<<<B_TOK / ROWS, 256, 0, stream>>>(xa, wb, b_enc, W_dec, b_dec, out);
}
